// CosFreqEncoding_68659347194252
// MI455X (gfx1250) — matrix operand order repaired and checked by reading, not compiled, not run on hardware
//
#include <hip/hip_runtime.h>
#include <hip/hip_bf16.h>

// MI455X / gfx1250, wave32. Core math: v_wmma_f32_16x16x32_bf16.
// out = x @ (W^T @ cos_basis), then out /= max(out).
//
// Workspace (~42.2 MB):
//   Wt  : bf16 [2048 x 2080]  = W^T, F padded 2074->2080 with zeros
//   Ct  : bf16 [2048 x 2080]  = cos_basis^T, padded
//   M2T : bf16 [2048 x 2048]  = (W^T @ cos_basis)^T   (B^T layout for stage 2)
//   xb  : bf16 [4096 x 2048]  = x converted to bf16
//   maxkey : uint32 ordered-float max accumulator

typedef __attribute__((ext_vector_type(16))) __bf16 v16bf;
typedef __attribute__((ext_vector_type(8)))  float  v8f;

union Frag { v16bf v; uint4 q[2]; };

__device__ __forceinline__ unsigned bf2pk(float lo, float hi) {
  // RNE fp32 -> bf16, packed pair
  unsigned a = __float_as_uint(lo);
  unsigned b = __float_as_uint(hi);
  a = (a + 0x7FFFu + ((a >> 16) & 1u)) >> 16;
  b = (b + 0x7FFFu + ((b >> 16) & 1u)) >> 16;
  return a | (b << 16);
}

// ---------------- prep A: fp32 (srcRows x srcCols) -> bf16 transpose (srcCols x dstLd), zero-padded
__global__ __launch_bounds__(1024)
void transpose_cvt_bf16(const float* __restrict__ src, unsigned short* __restrict__ dst,
                        int srcRows, int srcCols, int dstLd) {
  __shared__ float tile[32][33];
  const int tx = threadIdx.x, ty = threadIdx.y;
  const int srow = blockIdx.y * 32 + ty;   // F dimension (padded range)
  const int scol = blockIdx.x * 32 + tx;   // L dimension
  tile[ty][tx] = (srow < srcRows) ? src[(size_t)srow * srcCols + scol] : 0.0f;
  __syncthreads();
  const int drow = blockIdx.x * 32 + ty;   // L
  const int dcol = blockIdx.y * 32 + tx;   // F (grid covers exactly dstLd)
  const unsigned u = __float_as_uint(tile[tx][ty]);
  dst[(size_t)drow * dstLd + dcol] =
      (unsigned short)((u + 0x7FFFu + ((u >> 16) & 1u)) >> 16);
}

// ---------------- prep B: straight fp32 -> bf16 convert (4 elems/thread)
__global__ __launch_bounds__(256)
void cvt_f32_bf16(const float* __restrict__ src, unsigned short* __restrict__ dst, int n4) {
  const int i = blockIdx.x * blockDim.x + threadIdx.x;
  if (i < n4) {
    float4 f = ((const float4*)src)[i];
    ((uint2*)dst)[i] = make_uint2(bf2pk(f.x, f.y), bf2pk(f.z, f.w));
  }
}

// =====================================================================
// Shared GEMM core: 8 waves/block, each wave owns a 64x64 tile (4x4 WMMA
// tiles), block covers 128(M) x 256(N). 16 WMMAs per 16 b128 loads.
// A  : bf16 row-major (M x K), lane = row, chunks at K = hi*8 / hi*8+16
// BT : bf16 row-major (N x K), lane = column of B, chunks at K = hi*8 / hi*8+16
// =====================================================================
#define GEMM_PROLOG(LD_A, LD_B)                                            \
  const int lane = threadIdx.x & 31;                                       \
  const int wid  = threadIdx.x >> 5;                                       \
  const int l15 = lane & 15, hi = lane >> 4;                               \
  const int mb = blockIdx.x * 128 + (wid >> 2) * 64;                       \
  const int nb = blockIdx.y * 256 + (wid & 3)  * 64;                       \
  const unsigned short* ap[4];                                             \
  const unsigned short* bp[4];                                             \
  _Pragma("unroll")                                                        \
  for (int i = 0; i < 4; ++i) {                                            \
    ap[i] = A  + (size_t)(mb + i * 16 + l15) * (LD_A) + hi * 8;            \
    bp[i] = BT + (size_t)(nb + i * 16 + l15) * (LD_B) + hi * 8;            \
  }                                                                        \
  v8f acc[4][4] = {};                                                      \
  for (int k = 0; k < K; k += 32) {                                        \
    Frag a[4], b[4];                                                       \
    _Pragma("unroll")                                                      \
    for (int i = 0; i < 4; ++i) {                                          \
      a[i].q[0] = *(const uint4*)(ap[i] + k);                              \
      a[i].q[1] = *(const uint4*)(ap[i] + k + 16);                         \
      b[i].q[0] = *(const uint4*)(bp[i] + k);                              \
      b[i].q[1] = *(const uint4*)(bp[i] + k + 16);                         \
    }                                                                      \
    _Pragma("unroll")                                                      \
    for (int i = 0; i < 4; ++i)                                            \
      _Pragma("unroll")                                                    \
      for (int j = 0; j < 4; ++j)                                          \
        acc[i][j] = __builtin_amdgcn_wmma_f32_16x16x32_bf16(               \
            false, a[i].v, false, b[j].v, (short)0, acc[i][j], false, false); \
  }

// ---------------- stage 1: D = A x BT^T over K, store D transposed as bf16
__global__ __launch_bounds__(256)
void wmma_gemm_bf16_storeT(const unsigned short* __restrict__ A,   // Wt 2048 x 2080
                           const unsigned short* __restrict__ BT,  // Ct 2048 x 2080
                           unsigned short* __restrict__ DT,        // M2T 2048 x 2048
                           int K, int LD, int LDO) {
  GEMM_PROLOG(LD, LD)
  // D-frag: lane holds column n; VGPR v holds row m = base + v (+8 for hi lanes).
  // Transposed store makes each lane's 8 rows contiguous -> one 16B store/tile.
#pragma unroll
  for (int i = 0; i < 4; ++i)
#pragma unroll
    for (int j = 0; j < 4; ++j) {
      const v8f& c = acc[i][j];
      const int n = nb + j * 16 + l15;
      const int m = mb + i * 16 + hi * 8;
      uint4 p = make_uint4(bf2pk(c[0], c[1]), bf2pk(c[2], c[3]),
                           bf2pk(c[4], c[5]), bf2pk(c[6], c[7]));
      *(uint4*)(DT + (size_t)n * LDO + m) = p;
    }
}

// ---------------- stage 2: out(f32) = A x BT^T over K, fused global max
__global__ __launch_bounds__(256)
void wmma_gemm_bf16_f32out(const unsigned short* __restrict__ A,   // xb 4096 x 2048
                           const unsigned short* __restrict__ BT,  // M2T 2048 x 2048
                           float* __restrict__ out,                // 4096 x 2048
                           unsigned* __restrict__ maxkey,
                           int K, int LD, int LDO) {
  GEMM_PROLOG(LD, LD)
  float mx = -__builtin_inff();
#pragma unroll
  for (int i = 0; i < 4; ++i)
#pragma unroll
    for (int j = 0; j < 4; ++j) {
      const v8f& c = acc[i][j];
      const int n = nb + j * 16 + l15;
      const int m = mb + i * 16 + hi * 8;
#pragma unroll
      for (int v = 0; v < 8; ++v) {   // lanes 0-15 cover 16 contiguous n per store
        out[(size_t)(m + v) * LDO + n] = c[v];
        mx = fmaxf(mx, c[v]);
      }
    }
  for (int off = 16; off > 0; off >>= 1)
    mx = fmaxf(mx, __shfl_xor(mx, off, 32));
  if (lane == 0) {
    const unsigned b = __float_as_uint(mx);
    const unsigned key = b ^ ((b >> 31) ? 0xFFFFFFFFu : 0x80000000u);
    atomicMax(maxkey, key);
  }
}

// ---------------- final: out /= max(out)
__global__ __launch_bounds__(256)
void normalize_div(float* __restrict__ out, const unsigned* __restrict__ maxkey, int n4) {
  const unsigned k = *maxkey;
  const unsigned bits = (k & 0x80000000u) ? (k ^ 0x80000000u) : ~k;
  const float inv = 1.0f / __uint_as_float(bits);
  const int i = blockIdx.x * blockDim.x + threadIdx.x;
  if (i < n4) {
    float4 v = ((float4*)out)[i];
    v.x *= inv; v.y *= inv; v.z *= inv; v.w *= inv;
    ((float4*)out)[i] = v;
  }
}

extern "C" void kernel_launch(void* const* d_in, const int* in_sizes, int n_in,
                              void* d_out, int out_size, void* d_ws, size_t ws_size,
                              hipStream_t stream) {
  const float* x = (const float*)d_in[0];   // (4096, 2048)
  const float* W = (const float*)d_in[1];   // (2074, 2048)
  const float* C = (const float*)d_in[2];   // (2074, 2048)
  float* out = (float*)d_out;               // (4096, 2048)

  constexpr int B = 4096, L = 2048, F = 2074, Fp = 2080;

  char* ws = (char*)d_ws;
  size_t off = 0;
  unsigned short* Wt  = (unsigned short*)(ws + off); off += (size_t)L * Fp * 2;
  unsigned short* Ct  = (unsigned short*)(ws + off); off += (size_t)L * Fp * 2;
  unsigned short* M2T = (unsigned short*)(ws + off); off += (size_t)L * L * 2;
  unsigned short* xb  = (unsigned short*)(ws + off); off += (size_t)B * L * 2;
  unsigned* maxkey    = (unsigned*)(ws + off);

  hipMemsetAsync(maxkey, 0, sizeof(unsigned), stream);  // key 0 < any finite float key

  dim3 tb(32, 32);
  transpose_cvt_bf16<<<dim3(L / 32, Fp / 32), tb, 0, stream>>>(W, Wt, F, L, Fp);
  transpose_cvt_bf16<<<dim3(L / 32, Fp / 32), tb, 0, stream>>>(C, Ct, F, L, Fp);
  cvt_f32_bf16<<<(B * L / 4 + 255) / 256, 256, 0, stream>>>(x, xb, B * L / 4);

  // stage 1: M2T (2048x2048) = transpose(Wt x Ct^T), K = 2080
  wmma_gemm_bf16_storeT<<<dim3(L / 128, L / 256), 256, 0, stream>>>(Wt, Ct, M2T, Fp, Fp, L);

  // stage 2: out (4096x2048) = xb x M2, K = 2048, fused max
  wmma_gemm_bf16_f32out<<<dim3(B / 128, L / 256), 256, 0, stream>>>(xb, M2T, out, maxkey, L, L, L);

  normalize_div<<<(B * L / 4 + 255) / 256, 256, 0, stream>>>(out, maxkey, B * L / 4);
}
